// HypergraphDecoder_46059229282732
// MI455X (gfx1250) — matrix operand order repaired and checked by reading, not compiled, not run on hardware
//
#include <hip/hip_runtime.h>
#include <stdint.h>

// ---------------------------------------------------------------- constants
#define B_   2
#define N_   1024
#define E_   256
#define H_   64
#define D_   64
#define HM_  128
#define NEG_BIG 1.0e8f
#define BNN  (B_ * N_ * N_)

typedef __attribute__((ext_vector_type(16))) _Float16 v16h;
typedef __attribute__((ext_vector_type(8)))  float    v8f;

// ---------------------------------------------------------------- helpers
__device__ __forceinline__ float fast_tanh(float x) {
#if defined(__has_builtin)
#if __has_builtin(__builtin_amdgcn_tanhf)
    return __builtin_amdgcn_tanhf(x);          // v_tanh_f32 (gfx1250 TRANS)
#else
    float e = __expf(2.0f * x);
    return 1.0f - 2.0f / (e + 1.0f);
#endif
#else
    return tanhf(x);
#endif
}

__device__ __forceinline__ float stable_softplus(float x) {
    // softplus(x) = max(x,0) + log1p(exp(-|x|))
    return fmaxf(x, 0.0f) + log1pf(__expf(-fabsf(x)));
}

__device__ __forceinline__ uint32_t pcg_hash(uint32_t x) {
    x ^= x >> 16; x *= 0x7feb352du;
    x ^= x >> 15; x *= 0x846ca68bu;
    x ^= x >> 16;
    return x;
}

// ---------------------------------------------------------------- kernel 1
// dot[m, 0:64]  = encoder[m, :] @ wl          (m = b*N + n, flattened 2048 rows)
// dot[m,64:128] = encoder[m, :] @ wr
// One wave (32 threads) computes one 16x16 output tile with 8 chained
// v_wmma_f32_16x16x32_f16 ops (K = 256 = 8 * 32).
__global__ __launch_bounds__(32)
void proj_wmma_kernel(const float* __restrict__ X,   // [2048, 256]
                      const float* __restrict__ wl,  // [256, 64]
                      const float* __restrict__ wr,  // [256, 64]
                      float* __restrict__ dot)       // [2048, 128]
{
    const int lane = threadIdx.x;        // 0..31
    const int t    = blockIdx.x;         // 0..7  column tile (0-3 -> wl, 4-7 -> wr)
    const int mt   = blockIdx.y;         // 0..127 row tile
    const int half = lane >> 4;          // 0 or 1
    const int l15  = lane & 15;

    const float* W    = (t < 4) ? wl : wr;
    const int    ncol = ((t & 3) * 16) + l15;      // 0..63 local column in W
    const int    m    = mt * 16 + l15;             // A row for this lane
    const float* arow = X + (size_t)m * E_;

    v8f c = {};
    for (int kt = 0; kt < 8; ++kt) {
        const int k0 = kt * 32;

        // ---- A fragment: lane m=l15, k = half*8 + {0..7, 16..23} (+k0)
        const int ka = k0 + half * 8;
        const float4 x0 = *(const float4*)(arow + ka);
        const float4 x1 = *(const float4*)(arow + ka + 4);
        const float4 x2 = *(const float4*)(arow + ka + 16);
        const float4 x3 = *(const float4*)(arow + ka + 20);
        v16h a;
        a[0]  = (_Float16)x0.x; a[1]  = (_Float16)x0.y;
        a[2]  = (_Float16)x0.z; a[3]  = (_Float16)x0.w;
        a[4]  = (_Float16)x1.x; a[5]  = (_Float16)x1.y;
        a[6]  = (_Float16)x1.z; a[7]  = (_Float16)x1.w;
        a[8]  = (_Float16)x2.x; a[9]  = (_Float16)x2.y;
        a[10] = (_Float16)x2.z; a[11] = (_Float16)x2.w;
        a[12] = (_Float16)x3.x; a[13] = (_Float16)x3.y;
        a[14] = (_Float16)x3.z; a[15] = (_Float16)x3.w;

        // ---- B fragment: lane n=l15, k = half*8 + {0..7, 16..23} (+k0); W row-major [K,64]
        const int kb = k0 + half * 8;
        v16h bf;
        #pragma unroll
        for (int e = 0; e < 16; ++e)
            bf[e] = (_Float16)W[(size_t)(kb + (e & 7) + ((e >> 3) << 4)) * H_ + ncol];

        c = __builtin_amdgcn_wmma_f32_16x16x32_f16(
                /*neg_a=*/false, a, /*neg_b=*/false, bf,
                /*c_mod=*/(short)0, c, /*reuse_a=*/false, /*reuse_b=*/false);
    }

    // ---- D store: VGPR r -> row mt*16 + r + half*8, col t*16 + l15
    float* dcol = dot + (t * 16 + l15);
    #pragma unroll
    for (int r = 0; r < 8; ++r)
        dcol[(size_t)(mt * 16 + r + half * 8) * 128] = c[r];
}

// ---------------------------------------------------------------- kernel 2
// w[row] = softplus( relu(eq[row] @ W1 + b1) @ W2 + b2 ) + 1e-8
// One block (128 threads = one hidden unit each) per row.
__global__ __launch_bounds__(HM_)
void edge_mlp_kernel(const float* __restrict__ eq,   // [1024, 64]
                     const float* __restrict__ w1,   // [64, 128]
                     const float* __restrict__ b1,   // [128]
                     const float* __restrict__ w2,   // [128]
                     const float* __restrict__ b2,   // [1]
                     float* __restrict__ w_out)      // [1024]
{
    const int row = blockIdx.x;
    const int j   = threadIdx.x;                     // 0..127

    __shared__ float se[D_];
    __shared__ float red[HM_];
    if (j < D_) se[j] = eq[(size_t)row * D_ + j];
    __syncthreads();

    float acc = b1[j];
    #pragma unroll
    for (int d = 0; d < D_; ++d)
        acc = fmaf(se[d], w1[d * HM_ + j], acc);

    red[j] = fmaxf(acc, 0.0f) * w2[j];
    __syncthreads();
    #pragma unroll
    for (int s = HM_ / 2; s > 0; s >>= 1) {
        if (j < s) red[j] += red[j + s];
        __syncthreads();
    }
    if (j == 0) {
        float wl_ = red[0] + b2[0];
        w_out[row] = stable_softplus(wl_) + 1e-8f;
    }
}

// ---------------------------------------------------------------- kernel 3
// For each (b,i,j):
//   score  = sum_h u[h]*tanh(L[b,i,h]+R[b,j,h]) + l_bias - NEG_BIG*(i==j)
//   prob   = sigmoid(score); entropy = softplus(score) - score*prob
//   sample = bernoulli(prob) via counter-based hash
// One block per (b,i); 256 j's per block (gridDim.x = 4 chunks).
__global__ __launch_bounds__(256)
void pairwise_kernel(const float* __restrict__ dot,  // [2048, 128] (L | R)
                     const float* __restrict__ u,    // [64]
                     const float* __restrict__ lb,   // [1]
                     float* __restrict__ out_s,      // samples  [B,N,N]
                     float* __restrict__ out_m,      // scores   [B,N,N]
                     float* __restrict__ out_e)      // entropy  [B,N,N]
{
    const int bi  = blockIdx.y;                      // 0..2047 = b*N + i
    const int b   = bi >> 10;
    const int i   = bi & (N_ - 1);
    const int j   = blockIdx.x * 256 + threadIdx.x;  // 0..1023

    __shared__ float2 sLU[H_];                       // .x = L[h], .y = u[h]
    if (threadIdx.x < H_) {
        sLU[threadIdx.x].x = dot[(size_t)bi * 128 + threadIdx.x];
        sLU[threadIdx.x].y = u[threadIdx.x];
    }
    __syncthreads();

    // R row -> registers (64 floats, 16 x b128 loads)
    float rv[H_];
    const float4* rr = (const float4*)(dot + ((size_t)(b * N_ + j)) * 128 + H_);
    #pragma unroll
    for (int q = 0; q < 16; ++q) {
        float4 tq = rr[q];
        rv[4 * q + 0] = tq.x; rv[4 * q + 1] = tq.y;
        rv[4 * q + 2] = tq.z; rv[4 * q + 3] = tq.w;
    }

    float logit = 0.0f;
    #pragma unroll
    for (int h = 0; h < H_; ++h) {
        float2 luh = sLU[h];
        logit = fmaf(luh.y, fast_tanh(luh.x + rv[h]), logit);
    }

    float score = logit + lb[0];
    if (i == j) score -= NEG_BIG;

    float p   = 1.0f / (1.0f + __expf(-score));
    float ent = stable_softplus(score) - score * p;

    const uint32_t flat = ((uint32_t)bi << 10) | (uint32_t)j;
    float ur   = (float)(pcg_hash(flat) >> 8) * (1.0f / 16777216.0f);
    float smp  = (ur < p) ? 1.0f : 0.0f;

    const size_t gidx = (size_t)bi * N_ + j;
    out_s[gidx] = smp;
    out_m[gidx] = score;
    out_e[gidx] = ent;
}

// ---------------------------------------------------------------- launcher
extern "C" void kernel_launch(void* const* d_in, const int* in_sizes, int n_in,
                              void* d_out, int out_size, void* d_ws, size_t ws_size,
                              hipStream_t stream) {
    const float* enc   = (const float*)d_in[0];   // [2,1024,256]
    const float* wl    = (const float*)d_in[1];   // [256,64]
    const float* wr    = (const float*)d_in[2];   // [256,64]
    const float* u     = (const float*)d_in[3];   // [64]
    const float* lb    = (const float*)d_in[4];   // [1]
    const float* eq    = (const float*)d_in[5];   // [1024,64]
    const float* w1    = (const float*)d_in[6];   // [64,128]
    const float* b1    = (const float*)d_in[7];   // [128]
    const float* w2    = (const float*)d_in[8];   // [128,1]
    const float* b2    = (const float*)d_in[9];   // [1]

    float* out       = (float*)d_out;
    float* out_s     = out;                       // samples    [B,N,N]
    float* out_m     = out + (size_t)BNN;         // mask_scores[B,N,N]
    float* out_e     = out + (size_t)2 * BNN;     // entropy    [B,N,N]
    float* out_w     = out + (size_t)3 * BNN;     // w          [N]

    float* dot = (float*)d_ws;                    // [2048,128] = 1 MB scratch

    // 1) projections via WMMA: 8 col tiles x 128 row tiles, 1 wave each
    proj_wmma_kernel<<<dim3(8, 128), 32, 0, stream>>>(enc, wl, wr, dot);

    // 2) edge MLP -> w
    edge_mlp_kernel<<<N_, HM_, 0, stream>>>(eq, w1, b1, w2, b2, out_w);

    // 3) pairwise tanh-attention + epilogue
    pairwise_kernel<<<dim3(N_ / 256, B_ * N_), 256, 0, stream>>>(
        dot, u, lb, out_s, out_m, out_e);
}
